// Pointnet2MSG_63668595196429
// MI455X (gfx1250) — matrix operand order repaired and checked by reading, not compiled, not run on hardware
//
#include <hip/hip_runtime.h>
#include <cstdint>
#include <cstddef>

// ---------------------------------------------------------------------------
// CDNA5 (gfx1250) PointNet++ MSG forward.
// Activations & weights staged as bf16; every GEMM is a guard-free
// v_wmma_f32_16x16x32_bf16 stream with a 16x64 tile per wave:
//   per K-step: 2x b128 A-loads + 8x b128 B-loads + 4 WMMA, EXEC all-ones.
// A-operand is reused across 4 N-subtiles (4x less traffic on the dominant
// tall-skinny GEMMs of this bandwidth-bound network).
// ---------------------------------------------------------------------------

typedef unsigned short bf_t;                                  // raw bf16 bits
typedef __attribute__((ext_vector_type(16))) __bf16 v16bf;
typedef __attribute__((ext_vector_type(8)))  __bf16 v8bf;
typedef __attribute__((ext_vector_type(8)))  float  v8f;

#define CDIV(a, b) (((a) + (b) - 1) / (b))

__device__ __forceinline__ float bf2f(bf_t s) {
  union { unsigned u; float f; } x; x.u = ((unsigned)s) << 16; return x.f;
}
__device__ __forceinline__ bf_t f2bf(float f) {
  union { float f; unsigned u; } x; x.f = f;
  unsigned r = x.u + 0x7FFFu + ((x.u >> 16) & 1u);            // RNE
  return (bf_t)(r >> 16);
}

// ---------------------------------------------------------------------------
// Weight prep: f32 (N,K) -> bf16 (Np,Kp) zero-padded; scale/bias -> f32 (Np)
// with scale=0 in padding so padded output columns evaluate to exactly 0.
// ---------------------------------------------------------------------------
__global__ void prep_weights_kernel(const float* __restrict__ W, const float* __restrict__ s,
                                    const float* __restrict__ b, bf_t* __restrict__ Wp,
                                    float* __restrict__ sp, float* __restrict__ bp,
                                    int N, int K, int Np, int Kp) {
  int id = blockIdx.x * blockDim.x + threadIdx.x;
  int total = Np * Kp;
  if (id < total) {
    int k = id % Kp, n = id / Kp;
    float v = (n < N && k < K) ? W[(size_t)n * K + k] : 0.0f;
    Wp[id] = f2bf(v);
  } else if (id < total + 2 * Np) {
    int j = id - total;
    if (j < Np) sp[j] = (j < N) ? (s ? s[j] : 1.0f) : 0.0f;
    else        { int q = j - Np; bp[q] = (q < N && b) ? b[q] : 0.0f; }
  }
}

// ---------------------------------------------------------------------------
// WMMA GEMM (guard-free, 16x64 tile/wave): Y[m,n] = act(acc*S[n]+B[n]).
// X: (M,Kp) bf16, Wp: (Np,Kp) bf16, Y: (M,Np) bf16.
// M%16==0, Kp%32==0, Np%64==0 guaranteed by construction.
// ---------------------------------------------------------------------------
__global__ __launch_bounds__(32)
void gemm_wmma_bf16(const bf_t* __restrict__ Xu, const bf_t* __restrict__ Wu,
                    const float* __restrict__ Sp, const float* __restrict__ Bp,
                    bf_t* __restrict__ Yu, int Kp, int ldY, int relu_flag) {
  const __bf16* X = (const __bf16*)Xu;
  const __bf16* W = (const __bf16*)Wu;
  const int lane = threadIdx.x;
  const int g = lane >> 4;          // 0: lanes 0-15, 1: lanes 16-31
  const int lr = lane & 15;
  const long tileM = (long)blockIdx.x * 16;
  const int  tileN = blockIdx.y * 64;
  const __bf16* xrow  = X + (tileM + lr) * (long)Kp;          // A row
  const __bf16* wrow0 = W + (long)(tileN +  0 + lr) * Kp;     // B cols
  const __bf16* wrow1 = W + (long)(tileN + 16 + lr) * Kp;
  const __bf16* wrow2 = W + (long)(tileN + 32 + lr) * Kp;
  const __bf16* wrow3 = W + (long)(tileN + 48 + lr) * Kp;
  v8f acc0 = {}, acc1 = {}, acc2 = {}, acc3 = {};
  for (int k0 = 0; k0 < Kp; k0 += 32) {
    __builtin_prefetch(xrow + k0 + 128, 0, 1);                // global_prefetch_b8
    // A 16x32 fragment: e -> K=k0+16g+e
    v8bf alo = *(const v8bf*)(xrow + k0 + 16 * g);
    v8bf ahi = *(const v8bf*)(xrow + k0 + 16 * g + 8);
    v16bf afr;
#pragma unroll
    for (int e = 0; e < 8; ++e) { afr[e] = alo[e]; afr[e + 8] = ahi[e]; }
    // B 32x16 fragments: e -> K=k0+16g+e, N=lane&15 (32B-aligned v16bf loads)
    v16bf bf0 = *(const v16bf*)(wrow0 + k0 + 16 * g);
    v16bf bf1 = *(const v16bf*)(wrow1 + k0 + 16 * g);
    v16bf bf2 = *(const v16bf*)(wrow2 + k0 + 16 * g);
    v16bf bf3 = *(const v16bf*)(wrow3 + k0 + 16 * g);
    acc0 = __builtin_amdgcn_wmma_f32_16x16x32_bf16(false, afr, false, bf0, (short)0, acc0, false, false);
    acc1 = __builtin_amdgcn_wmma_f32_16x16x32_bf16(false, afr, false, bf1, (short)0, acc1, false, false);
    acc2 = __builtin_amdgcn_wmma_f32_16x16x32_bf16(false, afr, false, bf2, (short)0, acc2, false, false);
    acc3 = __builtin_amdgcn_wmma_f32_16x16x32_bf16(false, afr, false, bf3, (short)0, acc3, false, false);
  }
  // D layout: VGPR v holds M = v+8g, N = lane&15
  auto store_tile = [&](const v8f& a, int ncol0) {
    const int nst = ncol0 + lr;
    const float sv = Sp[nst];
    const float bv = Bp[nst];
#pragma unroll
    for (int v = 0; v < 8; ++v) {
      float y = a[v] * sv + bv;
      if (relu_flag) y = fmaxf(y, 0.0f);
      Yu[(tileM + v + 8 * g) * (long)ldY + nst] = f2bf(y);
    }
  };
  store_tile(acc0, tileN +  0);
  store_tile(acc1, tileN + 16);
  store_tile(acc2, tileN + 32);
  store_tile(acc3, tileN + 48);
}

// ---------------------------------------------------------------------------
// Data movement / pointwise kernels
// ---------------------------------------------------------------------------

__global__ void split_pc_kernel(const float* __restrict__ pc, float* __restrict__ xyz,
                                bf_t* __restrict__ feat, int BN) {
  int id = blockIdx.x * blockDim.x + threadIdx.x;
  if (id >= BN) return;
  const float* p = pc + (size_t)id * 6;
  xyz[(size_t)id * 3 + 0] = p[0];
  xyz[(size_t)id * 3 + 1] = p[1];
  xyz[(size_t)id * 3 + 2] = p[2];
  feat[(size_t)id * 3 + 0] = f2bf(p[3]);
  feat[(size_t)id * 3 + 1] = f2bf(p[4]);
  feat[(size_t)id * 3 + 2] = f2bf(p[5]);
}

// dst[r, dstOff+c] = src[r, c] (bf16), c in [0,nC)
__global__ void copy_block_bf_kernel(const bf_t* __restrict__ src, bf_t* __restrict__ dst,
                                     int rows, int srcLd, int dstLd, int dstOff, int nC) {
  int id = blockIdx.x * blockDim.x + threadIdx.x;
  if (id >= rows * nC) return;
  int c = id % nC, r = id / nC;
  dst[(size_t)r * dstLd + dstOff + c] = src[(size_t)r * srcLd + c];
}

// dst[r, dstOff+c] = (bf16)src_f32[r, c]
__global__ void copy_f2bf_kernel(const float* __restrict__ src, bf_t* __restrict__ dst,
                                 int rows, int srcLd, int dstLd, int dstOff, int nC) {
  int id = blockIdx.x * blockDim.x + threadIdx.x;
  if (id >= rows * nC) return;
  int c = id % nC, r = id / nC;
  dst[(size_t)r * dstLd + dstOff + c] = f2bf(src[(size_t)r * srcLd + c]);
}

// zero pad columns [c0, ld)
__global__ void zero_cols_kernel(bf_t* __restrict__ dst, int rows, int ld, int c0) {
  int nC = ld - c0;
  int id = blockIdx.x * blockDim.x + threadIdx.x;
  if (id >= rows * nC) return;
  int c = id % nC, r = id / nC;
  dst[(size_t)r * ld + c0 + c] = 0;
}

// dst[r, dstOff+c] = src[r/per, c]  (broadcast global feature)
__global__ void broadcast_rows_kernel(const bf_t* __restrict__ src, bf_t* __restrict__ dst,
                                      int rows, int per, int srcC, int dstLd, int dstOff) {
  int id = blockIdx.x * blockDim.x + threadIdx.x;
  if (id >= rows * srcC) return;
  int c = id % srcC, r = id / srcC;
  dst[(size_t)r * dstLd + dstOff + c] = src[(size_t)(r / per) * srcC + c];
}

// ---------------------------------------------------------------------------
// FPS (one block per batch, LDS argmax tree, jnp-scan semantics)
// ---------------------------------------------------------------------------
__global__ void fps_kernel(const float* __restrict__ xyz, float* __restrict__ dist,
                           int* __restrict__ fidx, int N, int npoint) {
  int b = blockIdx.x;
  const float* X = xyz + (size_t)b * N * 3;
  float* D = dist + (size_t)b * N;
  int* F = fidx + (size_t)b * npoint;
  __shared__ float rv[256];
  __shared__ int   ri[256];
  __shared__ int   sfar;
  int t = threadIdx.x;
  for (int n = t; n < N; n += blockDim.x) D[n] = 1e10f;
  if (t == 0) sfar = 0;
  __syncthreads();
  for (int it = 0; it < npoint; ++it) {
    int far = sfar;
    if (t == 0) F[it] = far;
    float cx = X[(size_t)far * 3 + 0];
    float cy = X[(size_t)far * 3 + 1];
    float cz = X[(size_t)far * 3 + 2];
    float bv = -1.0f; int bi = 0x7FFFFFFF;
    for (int n = t; n < N; n += blockDim.x) {
      float dx = X[(size_t)n * 3 + 0] - cx;
      float dy = X[(size_t)n * 3 + 1] - cy;
      float dz = X[(size_t)n * 3 + 2] - cz;
      float d = fminf(D[n], dx * dx + dy * dy + dz * dz);
      D[n] = d;
      if (d > bv) { bv = d; bi = n; }
    }
    rv[t] = bv; ri[t] = bi;
    __syncthreads();
    for (int s = blockDim.x / 2; s > 0; s >>= 1) {
      if (t < s) {
        if (rv[t + s] > rv[t] || (rv[t + s] == rv[t] && ri[t + s] < ri[t])) {
          rv[t] = rv[t + s]; ri[t] = ri[t + s];
        }
      }
      __syncthreads();
    }
    if (t == 0) sfar = ri[0];
    __syncthreads();
  }
}

__global__ void gather_xyz_kernel(const float* __restrict__ xyz, const int* __restrict__ fidx,
                                  float* __restrict__ out, int B, int npoint, int N) {
  int id = blockIdx.x * blockDim.x + threadIdx.x;
  if (id >= B * npoint) return;
  int b = id / npoint;
  int gi = fidx[id];
  const float* s = xyz + ((size_t)b * N + gi) * 3;
  out[(size_t)id * 3 + 0] = s[0];
  out[(size_t)id * 3 + 1] = s[1];
  out[(size_t)id * 3 + 2] = s[2];
}

// first K in-radius indices in index order; fill with first (or N-1)
__global__ void ball_query_kernel(const float* __restrict__ new_xyz,
                                  const float* __restrict__ cx, int* __restrict__ gidx,
                                  int B, int npoint, int N, float r2, int K) {
  int id = blockIdx.x * blockDim.x + threadIdx.x;
  if (id >= B * npoint) return;
  int b = id / npoint;
  const float* q = new_xyz + (size_t)id * 3;
  const float* P = cx + (size_t)b * N * 3;
  int* out = gidx + (size_t)id * K;
  float qx = q[0], qy = q[1], qz = q[2];
  int cnt = 0, first = N - 1;
  for (int n = 0; n < N && cnt < K; ++n) {
    float dx = P[(size_t)n * 3 + 0] - qx;
    float dy = P[(size_t)n * 3 + 1] - qy;
    float dz = P[(size_t)n * 3 + 2] - qz;
    if (dx * dx + dy * dy + dz * dz <= r2) { if (cnt == 0) first = n; out[cnt++] = n; }
  }
  for (int k = cnt; k < K; ++k) out[k] = first;
}

// grouped tensor (bf16, padded stride ld): [rel_xyz(3) | feats(C) | 0-pad]
__global__ void group_kernel(const float* __restrict__ cx, const bf_t* __restrict__ cf,
                             const float* __restrict__ new_xyz, const int* __restrict__ gidx,
                             bf_t* __restrict__ g, int B, int npoint, int N, int K,
                             int C, int ld) {
  size_t total = (size_t)B * npoint * K * ld;
  size_t id = (size_t)blockIdx.x * blockDim.x + threadIdx.x;
  if (id >= total) return;
  int c = (int)(id % ld);
  size_t r = id / ld;
  size_t bm = r / K;
  int b = (int)(bm / npoint);
  int gi = gidx[r];
  bf_t v;
  if (c < 3)          v = f2bf(cx[((size_t)b * N + gi) * 3 + c] - new_xyz[bm * 3 + c]);
  else if (c < 3 + C) v = cf[((size_t)b * N + gi) * C + (c - 3)];
  else                v = 0;
  g[id] = v;
}

// out[r, coff+c] = max_k in[r*K+k, c]
__global__ void maxpool_k_kernel(const bf_t* __restrict__ in, bf_t* __restrict__ out,
                                 int rows, int K, int C, int ldIn, int ldOut, int coff) {
  int id = blockIdx.x * blockDim.x + threadIdx.x;
  if (id >= rows * C) return;
  int c = id % C, r = id / C;
  const bf_t* p = in + (size_t)r * K * ldIn + c;
  float m = bf2f(p[0]);
  for (int k = 1; k < K; ++k) m = fmaxf(m, bf2f(p[(size_t)k * ldIn]));
  out[(size_t)r * ldOut + coff + c] = f2bf(m);
}

// out[b, c] = max_n in[b*Np+n, c]
__global__ void maxpool_pts_kernel(const bf_t* __restrict__ in, bf_t* __restrict__ out,
                                   int B, int Np, int C, int ldIn) {
  int id = blockIdx.x * blockDim.x + threadIdx.x;
  if (id >= B * C) return;
  int c = id % C, b = id / C;
  float m = bf2f(in[((size_t)b * Np) * ldIn + c]);
  for (int n = 1; n < Np; ++n) m = fmaxf(m, bf2f(in[((size_t)b * Np + n) * ldIn + c]));
  out[(size_t)b * C + c] = f2bf(m);
}

// 3-NN interpolation -> dst[row, dstOff..dstOff+C2) ; zero-pad [dstOff+C2, ld)
__global__ void three_interp_kernel(const float* __restrict__ xyz1, const float* __restrict__ xyz2,
                                    const bf_t* __restrict__ p2, bf_t* __restrict__ dst,
                                    int B, int N1, int N2, int C2, int ldP2, int ld, int dstOff) {
  int id = blockIdx.x * blockDim.x + threadIdx.x;
  if (id >= B * N1) return;
  int b = id / N1;
  const float* q = xyz1 + (size_t)id * 3;
  const float* P = xyz2 + (size_t)b * N2 * 3;
  float qx = q[0], qy = q[1], qz = q[2];
  float d0 = 1e30f, d1 = 1e30f, d2 = 1e30f;
  int i0 = 0, i1 = 0, i2 = 0;
  for (int n = 0; n < N2; ++n) {
    float dx = P[(size_t)n * 3 + 0] - qx;
    float dy = P[(size_t)n * 3 + 1] - qy;
    float dz = P[(size_t)n * 3 + 2] - qz;
    float d = dx * dx + dy * dy + dz * dz;
    if (d < d0)      { d2 = d1; i2 = i1; d1 = d0; i1 = i0; d0 = d; i0 = n; }
    else if (d < d1) { d2 = d1; i2 = i1; d1 = d;  i1 = n; }
    else if (d < d2) { d2 = d;  i2 = n; }
  }
  float w0 = 1.0f / (fmaxf(d0, 0.0f) + 1e-8f);
  float w1 = 1.0f / (fmaxf(d1, 0.0f) + 1e-8f);
  float w2 = 1.0f / (fmaxf(d2, 0.0f) + 1e-8f);
  float ws = w0 + w1 + w2;
  w0 /= ws; w1 /= ws; w2 /= ws;
  const bf_t* r0 = p2 + ((size_t)b * N2 + i0) * ldP2;
  const bf_t* r1 = p2 + ((size_t)b * N2 + i1) * ldP2;
  const bf_t* r2 = p2 + ((size_t)b * N2 + i2) * ldP2;
  bf_t* o = dst + (size_t)id * ld + dstOff;
  for (int c = 0; c < C2; ++c)
    o[c] = f2bf(w0 * bf2f(r0[c]) + w1 * bf2f(r1[c]) + w2 * bf2f(r2[c]));
  for (int c = dstOff + C2; c < ld; ++c) dst[(size_t)id * ld + c] = 0;
}

// tiny head: Cin -> 64 (relu, scale/bias) -> 3, L2-normalize. One block/batch.
__global__ void head_small_kernel(const bf_t* __restrict__ gf,
                                  const float* __restrict__ W1, const float* __restrict__ s1,
                                  const float* __restrict__ b1, const float* __restrict__ W2,
                                  const float* __restrict__ b2, float* __restrict__ out, int Cin) {
  int b = blockIdx.x;
  int t = threadIdx.x;  // 64 threads
  __shared__ float h[64];
  __shared__ float o[3];
  const bf_t* x = gf + (size_t)b * Cin;
  float acc = 0.0f;
  for (int c = 0; c < Cin; ++c) acc += bf2f(x[c]) * W1[(size_t)t * Cin + c];
  h[t] = fmaxf(acc * s1[t] + b1[t], 0.0f);
  __syncthreads();
  if (t < 3) {
    float a = b2[t];
    for (int j = 0; j < 64; ++j) a += h[j] * W2[t * 64 + j];
    o[t] = a;
  }
  __syncthreads();
  if (t == 0) {
    float nrm = fmaxf(sqrtf(o[0] * o[0] + o[1] * o[1] + o[2] * o[2]), 1e-12f);
    out[b * 3 + 0] = o[0] / nrm;
    out[b * 3 + 1] = o[1] / nrm;
    out[b * 3 + 2] = o[2] / nrm;
  }
}

// softmax over C=9 logits (bf16, stride ldL), write transposed (B, C, N) f32
__global__ void softmax_tr_kernel(const bf_t* __restrict__ logits, float* __restrict__ out,
                                  int B, int N, int C, int ldL) {
  int id = blockIdx.x * blockDim.x + threadIdx.x;
  if (id >= B * N) return;
  int b = id / N, n = id % N;
  const bf_t* l = logits + (size_t)id * ldL;
  float v[9];
  float mx = -1e30f;
  for (int c = 0; c < C; ++c) { v[c] = bf2f(l[c]); mx = fmaxf(mx, v[c]); }
  float s = 0.0f;
  for (int c = 0; c < C; ++c) { v[c] = __expf(v[c] - mx); s += v[c]; }
  float inv = 1.0f / s;
  for (int c = 0; c < C; ++c) out[((size_t)b * C + c) * N + n] = v[c] * inv;
}

// ---------------------------------------------------------------------------
// Host orchestration
// ---------------------------------------------------------------------------
extern "C" void kernel_launch(void* const* d_in, const int* in_sizes, int n_in,
                              void* d_out, int out_size, void* d_ws, size_t ws_size,
                              hipStream_t stream) {
  (void)in_sizes; (void)out_size; (void)ws_size;
  if (n_in < 190) return;

  const int B = 4, N0 = 16384;
  auto al32 = [](int x) { return (x + 31) & ~31; };
  auto al64 = [](int x) { return (x + 63) & ~63; };

  static const int   NPOINT[4]     = {1024, 256, 64, 16};
  static const float RADII[4][3]   = {{0.05f, 0.1f, 0.2f}, {0.1f, 0.2f, 0.3f},
                                      {0.2f, 0.4f, 0.6f}, {0.4f, 0.8f, 1.0f}};
  static const int   NSAMP[4][3]   = {{32, 64, 128}, {16, 32, 64}, {16, 32, 64}, {16, 32, 64}};
  static const int   MLPC[4][3][4] = {
      {{16, 16, 32, 64}, {32, 32, 64, 64}, {32, 32, 64, 128}},
      {{64, 64, 64, 128}, {64, 96, 96, 128}, {64, 96, 96, 256}},
      {{128, 196, 196, 256}, {128, 196, 196, 256}, {128, 196, 196, 512}},
      {{256, 256, 256, 512}, {256, 384, 384, 512}, {256, 384, 384, 1024}}};
  static const int   FPMID[4][2]   = {{128, 128}, {256, 256}, {512, 512}, {512, 512}};

  // --- raw parameter pointers (setup_inputs insertion order) ----------------
  int p = 0;
  const float* pc = (const float*)d_in[p++];
  struct L3 { const float *W, *s, *b; };
  L3 sa[4][3][4];
  for (int l = 0; l < 4; ++l)
    for (int sc = 0; sc < 3; ++sc)
      for (int y = 0; y < 4; ++y) {
        sa[l][sc][y].W = (const float*)d_in[p++];
        sa[l][sc][y].s = (const float*)d_in[p++];
        sa[l][sc][y].b = (const float*)d_in[p++];
      }
  L3 gsaL[2];
  for (int y = 0; y < 2; ++y) {
    gsaL[y].W = (const float*)d_in[p++];
    gsaL[y].s = (const float*)d_in[p++];
    gsaL[y].b = (const float*)d_in[p++];
  }
  L3 fpl[4][2];
  for (int i = 0; i < 4; ++i)
    for (int y = 0; y < 2; ++y) {
      fpl[i][y].W = (const float*)d_in[p++];
      fpl[i][y].s = (const float*)d_in[p++];
      fpl[i][y].b = (const float*)d_in[p++];
    }
  struct Head { L3 l1; const float *W2, *b2; };
  Head heads[3];  // up, fwd, seg
  for (int h = 0; h < 3; ++h) {
    heads[h].l1.W = (const float*)d_in[p++];
    heads[h].l1.s = (const float*)d_in[p++];
    heads[h].l1.b = (const float*)d_in[p++];
    heads[h].W2   = (const float*)d_in[p++];
    heads[h].b2   = (const float*)d_in[p++];
  }

  // --- workspace (byte allocator, 256B-aligned buffers) ---------------------
  char* base = (char*)d_ws;
  size_t off = 0;
  auto alloc = [&](size_t bytes) -> void* {
    void* q = base + off;
    off += (bytes + 255) & ~(size_t)255;
    return q;
  };
  float* xyz0  = (float*)alloc((size_t)B * N0 * 3 * 4);
  float* dist  = (float*)alloc((size_t)B * N0 * 4);
  int*   fidx  = (int*)alloc((size_t)B * 1024 * 4);
  float* lxyz1 = (float*)alloc((size_t)B * 1024 * 3 * 4);
  float* lxyz2 = (float*)alloc((size_t)B * 256 * 3 * 4);
  float* lxyz3 = (float*)alloc((size_t)B * 64 * 3 * 4);
  float* lxyz4 = (float*)alloc((size_t)B * 16 * 3 * 4);
  bf_t*  feat0 = (bf_t*)alloc((size_t)B * N0 * 3 * 2);
  bf_t*  lf1   = (bf_t*)alloc((size_t)B * 1024 * 256 * 2);
  bf_t*  lf2   = (bf_t*)alloc((size_t)B * 256 * 512 * 2);
  bf_t*  lf3   = (bf_t*)alloc((size_t)B * 64 * 1024 * 2);
  bf_t*  lf4   = (bf_t*)alloc((size_t)B * 16 * 2048 * 2);
  bf_t*  lf0n  = (bf_t*)alloc((size_t)B * N0 * 128 * 2);
  bf_t*  gfeat = (bf_t*)alloc((size_t)B * 512 * 2);
  int*   gidx  = (int*)alloc((size_t)B * 1024 * 128 * 4);
  bf_t*  gbuf  = (bf_t*)alloc((size_t)19000000 * 2);   // max 65536 x 288
  bf_t*  bufA  = (bf_t*)alloc((size_t)42000000 * 2);   // max 65536 x 640
  bf_t*  bufB  = (bf_t*)alloc((size_t)33600000 * 2);   // max 524288 x 64
  bf_t*  wpad  = (bf_t*)alloc((size_t)12000000 * 2);   // padded bf16 weights
  float* sbpad = (float*)alloc((size_t)160000 * 4);    // padded scale/bias

  // --- prep all GEMM layers (padded bf16 weights + f32 scale/bias) ----------
  // Kp = stride of the layer's input buffer; Np = al64(N) so the 16x64-tile
  // GEMM needs no N guards (padded cols compute exactly 0 via scale=0).
  struct PLayer { const bf_t* W; const float* s; const float* b; int Kp; int Np; };
  size_t woff = 0, sboff = 0;
  auto prep = [&](const float* W, const float* s, const float* b,
                  int N, int K, int Kp) -> PLayer {
    int Np = al64(N);
    bf_t* Wp = wpad + woff;  woff += (size_t)Np * Kp;
    float* sp = sbpad + sboff; sboff += Np;
    float* bp = sbpad + sboff; sboff += Np;
    int total = Np * Kp + 2 * Np;
    prep_weights_kernel<<<CDIV(total, 256), 256, 0, stream>>>(W, s, b, Wp, sp, bp, N, K, Np, Kp);
    return PLayer{Wp, sp, bp, Kp, Np};
  };

  PLayer psa[4][3][4];
  {
    int cin0[4] = {3, 256, 512, 1024};
    for (int l = 0; l < 4; ++l)
      for (int sc = 0; sc < 3; ++sc) {
        int cin = cin0[l] + 3;
        int kp = al32(cin);                       // grouped buffer stride
        for (int y = 0; y < 4; ++y) {
          psa[l][sc][y] = prep(sa[l][sc][y].W, sa[l][sc][y].s, sa[l][sc][y].b,
                               MLPC[l][sc][y], cin, kp);
          cin = MLPC[l][sc][y];
          kp = psa[l][sc][y].Np;                  // chained stride
        }
      }
  }
  PLayer pgsa[2];
  pgsa[0] = prep(gsaL[0].W, gsaL[0].s, gsaL[0].b, 512, 2051, al32(2051));
  pgsa[1] = prep(gsaL[1].W, gsaL[1].s, gsaL[1].b, 512, 512, pgsa[0].Np);
  PLayer pfp[4][2];
  {
    int fpcin[4] = {259, 768, 1024, 3072};
    for (int i = 0; i < 4; ++i) {
      pfp[i][0] = prep(fpl[i][0].W, fpl[i][0].s, fpl[i][0].b, FPMID[i][0], fpcin[i], al32(fpcin[i]));
      pfp[i][1] = prep(fpl[i][1].W, fpl[i][1].s, fpl[i][1].b, FPMID[i][1], FPMID[i][0], pfp[i][0].Np);
    }
  }
  PLayer pseg1 = prep(heads[2].l1.W, heads[2].l1.s, heads[2].l1.b, 64, 640, 640);
  PLayer pseg2 = prep(heads[2].W2, nullptr, heads[2].b2, 9, 64, pseg1.Np);

  auto gemm = [&](const bf_t* X, const PLayer& L, bf_t* Y, int M, int relu) {
    dim3 grid(M / 16, L.Np / 64);
    gemm_wmma_bf16<<<grid, 32, 0, stream>>>(X, L.W, L.s, L.b, Y, L.Kp, L.Np, relu);
  };

  // --- split pointcloud -----------------------------------------------------
  split_pc_kernel<<<CDIV(B * N0, 256), 256, 0, stream>>>(pc, xyz0, feat0, B * N0);

  // --- SA levels ------------------------------------------------------------
  float* lxyz_arr[5]  = {xyz0, lxyz1, lxyz2, lxyz3, lxyz4};
  bf_t*  lfeat_arr[5] = {feat0, lf1, lf2, lf3, lf4};
  int lN[5] = {N0, 1024, 256, 64, 16};
  int lC[5]; lC[0] = 3;

  const float* cur_xyz = xyz0;
  const bf_t*  cur_feat = feat0;
  int curN = N0, curC = 3;

  for (int lvl = 0; lvl < 4; ++lvl) {
    int np = NPOINT[lvl];
    fps_kernel<<<B, 256, 0, stream>>>(cur_xyz, dist, fidx, curN, np);
    gather_xyz_kernel<<<CDIV(B * np, 256), 256, 0, stream>>>(cur_xyz, fidx,
                                                             lxyz_arr[lvl + 1], B, np, curN);
    int outC = MLPC[lvl][0][3] + MLPC[lvl][1][3] + MLPC[lvl][2][3];
    bf_t* lf = lfeat_arr[lvl + 1];
    int coff = 0;
    for (int sc = 0; sc < 3; ++sc) {
      int K = NSAMP[lvl][sc];
      float r = RADII[lvl][sc];
      ball_query_kernel<<<CDIV(B * np, 128), 128, 0, stream>>>(
          lxyz_arr[lvl + 1], cur_xyz, gidx, B, np, curN, r * r, K);
      int ldG = al32(curC + 3);
      size_t gtotal = (size_t)B * np * K * ldG;
      group_kernel<<<(int)CDIV(gtotal, (size_t)256), 256, 0, stream>>>(
          cur_xyz, cur_feat, lxyz_arr[lvl + 1], gidx, gbuf, B, np, curN, K, curC, ldG);
      int rows = B * np * K;        // always a multiple of 16
      const bf_t* xin = gbuf;
      bf_t* pingpong[2] = {bufA, bufB};
      int cout = 0;
      for (int y = 0; y < 4; ++y) {
        bf_t* yo = pingpong[y & 1];
        gemm(xin, psa[lvl][sc][y], yo, rows, /*relu=*/1);
        xin = yo;
        cout = MLPC[lvl][sc][y];
      }
      maxpool_k_kernel<<<CDIV(B * np * cout, 256), 256, 0, stream>>>(
          xin, lf, B * np, K, cout, psa[lvl][sc][3].Np, outC, coff);
      coff += cout;
    }
    cur_xyz = lxyz_arr[lvl + 1];
    cur_feat = lf;
    curN = np;
    curC = outC;
    lC[lvl + 1] = outC;
  }

  // --- global SA ------------------------------------------------------------
  int rows4 = B * 16;
  int ldGsa = al32(2051);  // 2080
  copy_f2bf_kernel<<<CDIV(rows4 * 3, 256), 256, 0, stream>>>(lxyz4, bufA, rows4, 3, ldGsa, 0, 3);
  copy_block_bf_kernel<<<CDIV(rows4 * 2048, 256), 256, 0, stream>>>(lf4, bufA, rows4, 2048, ldGsa, 3, 2048);
  zero_cols_kernel<<<CDIV(rows4 * (ldGsa - 2051), 256), 256, 0, stream>>>(bufA, rows4, ldGsa, 2051);
  gemm(bufA, pgsa[0], bufB, rows4, 1);
  gemm(bufB, pgsa[1], bufA, rows4, 1);
  maxpool_pts_kernel<<<CDIV(B * 512, 256), 256, 0, stream>>>(bufA, gfeat, B, 16, 512, pgsa[1].Np);

  // --- heads up / fwd -------------------------------------------------------
  float* out = (float*)d_out;
  head_small_kernel<<<B, 64, 0, stream>>>(gfeat, heads[0].l1.W, heads[0].l1.s, heads[0].l1.b,
                                          heads[0].W2, heads[0].b2, out + 0, 512);
  head_small_kernel<<<B, 64, 0, stream>>>(gfeat, heads[1].l1.W, heads[1].l1.s, heads[1].l1.b,
                                          heads[1].W2, heads[1].b2, out + B * 3, 512);

  // --- feature propagation (i = 3..0) ---------------------------------------
  for (int i = 3; i >= 0; --i) {
    int N1 = lN[i], N2 = lN[i + 1];
    int C1 = lC[i], C2 = lC[i + 1];
    int ldIn = al32(C1 + C2);
    int rows = B * N1;
    int srcLd = (i == 0) ? 3 : lC[i];   // SA feature buffers are tightly strided
    copy_block_bf_kernel<<<CDIV(rows * C1, 256), 256, 0, stream>>>(
        lfeat_arr[i], bufA, rows, srcLd, ldIn, 0, C1);
    three_interp_kernel<<<CDIV(rows, 128), 128, 0, stream>>>(
        lxyz_arr[i], lxyz_arr[i + 1], lfeat_arr[i + 1], bufA, B, N1, N2, C2, lC[i + 1], ldIn, C1);
    gemm(bufA, pfp[i][0], bufB, rows, 1);
    bf_t* dstf = (i == 0) ? lf0n : lfeat_arr[i];
    gemm(bufB, pfp[i][1], dstf, rows, 1);
    lfeat_arr[i] = dstf;
    lC[i] = FPMID[i][1];
  }

  // --- segmentation head ----------------------------------------------------
  int rowsN = B * N0;
  int segCin = lC[0] + 512;  // 640 (multiple of 64)
  copy_block_bf_kernel<<<CDIV(rowsN * lC[0], 256), 256, 0, stream>>>(
      lfeat_arr[0], bufA, rowsN, lC[0], segCin, 0, lC[0]);
  broadcast_rows_kernel<<<CDIV(rowsN * 512, 256), 256, 0, stream>>>(
      gfeat, bufA, rowsN, N0, 512, segCin, lC[0]);
  gemm(bufA, pseg1, bufB, rowsN, 1);
  gemm(bufB, pseg2, gbuf, rowsN, 0);   // logits, stride pseg2.Np (=64)
  softmax_tr_kernel<<<CDIV(rowsN, 128), 128, 0, stream>>>(gbuf, out + 24, B, N0, 9, pseg2.Np);
}
